// BlockAttnResTransformerBlock_62328565400278
// MI455X (gfx1250) — matrix operand order repaired and checked by reading, not compiled, not run on hardware
//
#include <hip/hip_runtime.h>
#include <hip/hip_bf16.h>

// ---------------------------------------------------------------------------
// MI455X (gfx1250, wave32) implementation.
//   dims: N=8 completed blocks, B=4, T=2048, D=1024  -> ROWS = B*T = 8192
// ---------------------------------------------------------------------------

typedef __attribute__((ext_vector_type(4)))  _Float16     v4h;
typedef __attribute__((ext_vector_type(8)))  _Float16     v8h;
typedef __attribute__((ext_vector_type(16))) _Float16     v16h;
typedef __attribute__((ext_vector_type(8)))  float        v8f;
typedef __attribute__((ext_vector_type(4)))  unsigned int u32x4;
typedef __attribute__((ext_vector_type(8)))  int          i32x8;
typedef __attribute__((ext_vector_type(4)))  int          i32x4;

#define DIMD      1024
#define ROWS      8192
#define NBLK      9
#define EPSV      1e-6f
#define INV_D     (1.0f/1024.0f)
#define INV_SCALE (1.0f/32.0f)     // 1/sqrt(D)

#if __has_builtin(__builtin_amdgcn_tensor_load_to_lds) && \
    __has_builtin(__builtin_amdgcn_s_wait_tensorcnt)
#define USE_TDM 1
#else
#define USE_TDM 0
#endif

// ---------------------------------------------------------------------------
// Prep: fp32 weight matrix [1024x1024] -> fp16   (1024 blocks x 256 thr x 4)
// ---------------------------------------------------------------------------
__global__ __launch_bounds__(256) void prep_w_kernel(const float* __restrict__ w,
                                                     _Float16* __restrict__ out) {
    const int i = (blockIdx.x * 256 + threadIdx.x) * 4;
    float4 x = *(const float4*)(w + i);
    v4h o = { (_Float16)x.x, (_Float16)x.y, (_Float16)x.z, (_Float16)x.w };
    *(v4h*)(out + i) = o;
}

// Prep: qn = q * res_norm_w (folds rmsnorm weight into the query)  [1024]
__global__ __launch_bounds__(256) void prep_qn_kernel(const float* __restrict__ q,
                                                      const float* __restrict__ nw,
                                                      float* __restrict__ qn) {
    const int i = threadIdx.x * 4;
    float4 a = *(const float4*)(q + i);
    float4 b = *(const float4*)(nw + i);
    float4 o = { a.x * b.x, a.y * b.y, a.z * b.z, a.w * b.w };
    *(float4*)(qn + i) = o;
}

// ---------------------------------------------------------------------------
// Attention stage: one workgroup (256 thr = 8 waves) per (b,t) row.
// Each thread holds 9 rows x float4 in registers -> V read from HBM once.
// Produces A[r, :] = rmsnorm(h, out_norm_w) in fp16 (GEMM LHS).
// ---------------------------------------------------------------------------
__global__ __launch_bounds__(256) void attn_stage_kernel(
    const float* __restrict__ Vc,     // [8,4,2048,1024] completed blocks
    const float* __restrict__ Vp,     // [4,2048,1024]   partial (per sub-layer)
    const float* __restrict__ qn,     // [1024] q * res_norm_w
    const float* __restrict__ onw,    // [1024] output rmsnorm weight
    _Float16* __restrict__ Aout)      // [8192,1024] fp16
{
    const int r    = blockIdx.x;          // 0..8191
    const int tid  = threadIdx.x;
    const int lane = tid & 31;
    const int wid  = tid >> 5;
    const int b    = r >> 11;
    const int t    = r & 2047;
    const int c4   = tid << 2;            // 4 floats per thread covers D=1024

    __shared__ float red[8][19];

    const float4 q4 = *(const float4*)(qn + c4);

    float4 v[NBLK];
#pragma unroll
    for (int n = 0; n < 8; ++n) {
        const size_t off = ((((size_t)n * 4 + b) * 2048 + t) * 1024) + c4;
        v[n] = *(const float4*)(Vc + off);
    }
    v[8] = *(const float4*)(Vp + (size_t)r * 1024 + c4);

    float ss[NBLK], dt[NBLK];
#pragma unroll
    for (int n = 0; n < NBLK; ++n) {
        const float4 x = v[n];
        ss[n] = x.x*x.x + x.y*x.y + x.z*x.z + x.w*x.w;
        dt[n] = x.x*q4.x + x.y*q4.y + x.z*q4.z + x.w*q4.w;
    }

    // wave32 butterfly reduction, then cross-wave via LDS
#pragma unroll
    for (int off = 16; off > 0; off >>= 1) {
#pragma unroll
        for (int n = 0; n < NBLK; ++n) {
            ss[n] += __shfl_xor(ss[n], off, 32);
            dt[n] += __shfl_xor(dt[n], off, 32);
        }
    }
    if (lane == 0) {
#pragma unroll
        for (int n = 0; n < NBLK; ++n) { red[wid][n] = ss[n]; red[wid][9 + n] = dt[n]; }
    }
    __syncthreads();

    float alpha[NBLK];
    {
        float l[NBLK], m = -3.4e38f;
#pragma unroll
        for (int n = 0; n < NBLK; ++n) {
            float s = 0.f, d = 0.f;
#pragma unroll
            for (int w = 0; w < 8; ++w) { s += red[w][n]; d += red[w][9 + n]; }
            const float rms = rsqrtf(s * INV_D + EPSV);
            l[n] = d * rms * INV_SCALE;      // dot(q, rmsnorm(V)) / sqrt(D)
            m = fmaxf(m, l[n]);
        }
        float sum = 0.f;
#pragma unroll
        for (int n = 0; n < NBLK; ++n) { alpha[n] = __expf(l[n] - m); sum += alpha[n]; }
        const float inv = 1.0f / sum;
#pragma unroll
        for (int n = 0; n < NBLK; ++n) alpha[n] *= inv;
    }

    float4 h = {0.f, 0.f, 0.f, 0.f};
#pragma unroll
    for (int n = 0; n < NBLK; ++n) {
        h.x += alpha[n] * v[n].x;  h.y += alpha[n] * v[n].y;
        h.z += alpha[n] * v[n].z;  h.w += alpha[n] * v[n].w;
    }

    // rmsnorm(h) -> fp16 GEMM operand
    float sh = h.x*h.x + h.y*h.y + h.z*h.z + h.w*h.w;
#pragma unroll
    for (int off = 16; off > 0; off >>= 1) sh += __shfl_xor(sh, off, 32);
    if (lane == 0) red[wid][18] = sh;
    __syncthreads();
    float st = 0.f;
#pragma unroll
    for (int w = 0; w < 8; ++w) st += red[w][18];
    const float rms = rsqrtf(st * INV_D + EPSV);

    const float4 w4 = *(const float4*)(onw + c4);
    v4h o = { (_Float16)(h.x * rms * w4.x), (_Float16)(h.y * rms * w4.y),
              (_Float16)(h.z * rms * w4.z), (_Float16)(h.w * rms * w4.w) };
    *(v4h*)(Aout + (size_t)r * 1024 + c4) = o;
}

// ---------------------------------------------------------------------------
// GEMM: dst = res + A @ W^T     A:[8192,1024] f16, W:[1024,1024] f16 row-major
// Block tile 128x128, 8 waves, wave tile 64x32 = 4x2 wmma 16x16x32_f16 tiles.
// Tile feed: TDM tensor_load_to_lds (double-buffered, hardware row padding),
// fallback: manual global->LDS staging. Fragment packing per ISA §7.12.2.
// ---------------------------------------------------------------------------
#define BM   128
#define BN   128
#define BK   32
#define LSTR 40   // padded LDS row stride in halves (80B = 64B row + 16B pad)

#if USE_TDM
// Issue one TDM 2D tile load: 128 rows x 32 halves from row-major [rows,1024]
// f16 tensor at gsrc (tile origin), into LDS at lds_addr with 16B row padding.
// D# layout per CDNA5 ISA §8.3 (group0) / §8.4 (group1).
__device__ __forceinline__ void tdm_load_tile(const _Float16* gsrc,
                                              unsigned lds_addr,
                                              unsigned tensor_rows) {
    const unsigned long long ga = (unsigned long long)(size_t)gsrc;
    u32x4 g0;
    g0.x = 1u;                                            // count=1, user mode
    g0.y = lds_addr;                                      // lds_addr [63:32]
    g0.z = (unsigned)ga;                                  // global_addr lo
    g0.w = (unsigned)((ga >> 32) & 0x01ffffffu) | (2u << 30);  // ga hi | type=2
    i32x8 g1;
    g1[0] = (int)((1u << 16)   |   // data_size = 1 -> 2 bytes
                  (1u << 20)   |   // pad_enable
                  (3u << 22)   |   // pad_interval: 16 DWORDs (one 64B row)
                  (3u << 25));     // pad_amount:   4 DWORDs (16B) -> 80B stride
    g1[1] = (int)(1024u << 16);                  // tensor_dim0 = 1024 elems
    g1[2] = (int)((tensor_rows & 0xffffu) << 16);// tensor_dim1 lo
    g1[3] = (int)(((tensor_rows >> 16) & 0xffffu) | (32u << 16)); // dim1 hi | tile_dim0=32
    g1[4] = (int)128u;                           // tile_dim1 = 128, tile_dim2 = 0
    g1[5] = (int)1024u;                          // tensor_dim0_stride = 1024
    g1[6] = 0;
    g1[7] = 0;
    const i32x4 z4 = {0, 0, 0, 0};
#if defined(__clang_major__) && (__clang_major__ >= 23)
    const i32x8 z8 = {0, 0, 0, 0, 0, 0, 0, 0};
    __builtin_amdgcn_tensor_load_to_lds(g0, g1, z4, z4, z8, 0);
#else
    __builtin_amdgcn_tensor_load_to_lds(g0, g1, z4, z4, 0);
#endif
}
#endif

__global__ __launch_bounds__(256) void gemm_f16_wmma_kernel(
    const _Float16* __restrict__ A,
    const _Float16* __restrict__ W,
    const float* __restrict__ res,
    float* __restrict__ dst)
{
    __shared__ _Float16 As[2][BM * LSTR];
    __shared__ _Float16 Bs[2][BN * LSTR];

    const int tid  = threadIdx.x;
    const int lane = tid & 31;
    const int wid  = tid >> 5;
    const int m0   = blockIdx.x * BM;
    const int n0   = blockIdx.y * BN;

    const int wm = (wid & 1) * 64;    // wave M offset in block tile
    const int wn = (wid >> 1) * 32;   // wave N offset in block tile

    v8f acc[4][2] = {};

    const int hal  = lane & 15;
    const int aLo  = (lane < 16) ? 0 : 8;    // A frag: K 0..7 / 8..15, then +16
    const int bCol = (lane < 16) ? 0 : 8;    // B frag: K 0..7 / 8..15, then +16

#if USE_TDM
    const unsigned ldsA0 = (unsigned)(size_t)&As[0][0];
    const unsigned ldsA1 = (unsigned)(size_t)&As[1][0];
    const unsigned ldsB0 = (unsigned)(size_t)&Bs[0][0];
    const unsigned ldsB1 = (unsigned)(size_t)&Bs[1][0];
    const _Float16* tileA = A + (size_t)m0 * DIMD;   // tile origin row m0, k 0
    const _Float16* tileB = W + (size_t)n0 * DIMD;   // tile origin row n0, k 0

    if (wid == 0) {                                  // prologue: k0 = 0 tiles
        tdm_load_tile(tileA, ldsA0, (unsigned)ROWS);
        tdm_load_tile(tileB, ldsB0, (unsigned)DIMD);
    }
#else
    const int lrow = tid >> 1;
    const int lcol = (tid & 1) * 16;
    const _Float16* gA = A + (size_t)(m0 + lrow) * DIMD + lcol;
    const _Float16* gB = W + (size_t)(n0 + lrow) * DIMD + lcol;
#endif

    int buf = 0;
    for (int k0 = 0; k0 < DIMD; k0 += BK) {
#if USE_TDM
        if (wid == 0) {
            if (k0 + BK < DIMD) {       // stage k0+BK into the other buffer
                const unsigned la = buf ? ldsA0 : ldsA1;
                const unsigned lb = buf ? ldsB0 : ldsB1;
                tdm_load_tile(tileA + k0 + BK, la, (unsigned)ROWS);
                tdm_load_tile(tileB + k0 + BK, lb, (unsigned)DIMD);
                // TDM ops from one wave retire in order: <=2 outstanding
                // means the current buffer's pair has landed in LDS.
                __builtin_amdgcn_s_wait_tensorcnt(2);
            } else {
                __builtin_amdgcn_s_wait_tensorcnt(0);
            }
        }
        __syncthreads();
#else
        v8h a0 = *(const v8h*)(gA + k0);
        v8h a1 = *(const v8h*)(gA + k0 + 8);
        v8h b0 = *(const v8h*)(gB + k0);
        v8h b1 = *(const v8h*)(gB + k0 + 8);
        if (k0 + BK < DIMD) {
            __builtin_prefetch(gA + k0 + BK, 0, 0);
            __builtin_prefetch(gB + k0 + BK, 0, 0);
        }
        *(v8h*)&As[buf][lrow * LSTR + lcol]     = a0;
        *(v8h*)&As[buf][lrow * LSTR + lcol + 8] = a1;
        *(v8h*)&Bs[buf][lrow * LSTR + lcol]     = b0;
        *(v8h*)&Bs[buf][lrow * LSTR + lcol + 8] = b1;
        __syncthreads();
#endif

        v16h af[4], bf[2];
#pragma unroll
        for (int mi = 0; mi < 4; ++mi) {
            const _Float16* p = &As[buf][(wm + mi * 16 + hal) * LSTR];
            v8h lo = *(const v8h*)(p + aLo);
            v8h hi = *(const v8h*)(p + aLo + 16);
            af[mi] = __builtin_shufflevector(lo, hi, 0,1,2,3,4,5,6,7,8,9,10,11,12,13,14,15);
        }
#pragma unroll
        for (int ni = 0; ni < 2; ++ni) {
            const _Float16* p = &Bs[buf][(wn + ni * 16 + hal) * LSTR + bCol];
            v8h lo = *(const v8h*)p;
            v8h hi = *(const v8h*)(p + 16);
            bf[ni] = __builtin_shufflevector(lo, hi, 0,1,2,3,4,5,6,7,8,9,10,11,12,13,14,15);
        }
#pragma unroll
        for (int mi = 0; mi < 4; ++mi)
#pragma unroll
            for (int ni = 0; ni < 2; ++ni)
                acc[mi][ni] = __builtin_amdgcn_wmma_f32_16x16x32_f16(
                    false, af[mi], false, bf[ni], (short)0, acc[mi][ni], false, false);
        __syncthreads();
        buf ^= 1;
    }

    // epilogue: C layout (§7.12.2): (vgpr rr, lane l) -> M = rr + (l>=16?8:0), N = l&15
    const int nl  = lane & 15;
    const int rof = (lane >> 4) * 8;
#pragma unroll
    for (int mi = 0; mi < 4; ++mi)
#pragma unroll
        for (int ni = 0; ni < 2; ++ni) {
            const size_t col  = (size_t)(n0 + wn + ni * 16 + nl);
            const size_t rowb = (size_t)(m0 + wm + mi * 16 + rof);
            float rv[8];
#pragma unroll
            for (int rr = 0; rr < 8; ++rr)            // batch residual loads
                rv[rr] = res[(rowb + rr) * DIMD + col];
#pragma unroll
            for (int rr = 0; rr < 8; ++rr)
                dst[(rowb + rr) * DIMD + col] = rv[rr] + acc[mi][ni][rr];
        }
}

// ---------------------------------------------------------------------------
// Launch
// ---------------------------------------------------------------------------
extern "C" void kernel_launch(void* const* d_in, const int* in_sizes, int n_in,
                              void* d_out, int out_size, void* d_ws, size_t ws_size,
                              hipStream_t stream) {
    (void)in_sizes; (void)n_in; (void)out_size; (void)ws_size;

    const float* Vc   = (const float*)d_in[0];   // completed_blocks [8,4,2048,1024]
    const float* Pb   = (const float*)d_in[1];   // partial_block   [4,2048,1024]
    const float* anw  = (const float*)d_in[2];   // attn_norm_w
    const float* aw   = (const float*)d_in[3];   // attn_w [1024,1024]
    const float* mnw  = (const float*)d_in[4];   // mlp_norm_w
    const float* mw   = (const float*)d_in[5];   // mlp_w [1024,1024]
    const float* q1   = (const float*)d_in[6];   // attn_res_query
    const float* rnw1 = (const float*)d_in[7];   // attn_res_norm_w
    const float* q2   = (const float*)d_in[8];   // mlp_res_query
    const float* rnw2 = (const float*)d_in[9];   // mlp_res_norm_w
    float* out = (float*)d_out;                  // [4,2048,1024]

    char* ws = (char*)d_ws;
    _Float16* Abuf = (_Float16*)ws;                               // 16 MB
    _Float16* W1h  = (_Float16*)(ws + (size_t)16 * 1024 * 1024);  //  2 MB
    _Float16* W2h  = (_Float16*)(ws + (size_t)18 * 1024 * 1024);  //  2 MB
    float*    qn1  = (float*)   (ws + (size_t)20 * 1024 * 1024);  //  4 KB
    float*    qn2  = qn1 + DIMD;

    // preprocessing
    prep_w_kernel<<<1024, 256, 0, stream>>>(aw, W1h);
    prep_w_kernel<<<1024, 256, 0, stream>>>(mw, W2h);
    prep_qn_kernel<<<1, 256, 0, stream>>>(q1, rnw1, qn1);
    prep_qn_kernel<<<1, 256, 0, stream>>>(q2, rnw2, qn2);

    const dim3 ggrid(ROWS / BM, DIMD / BN);

    // sub-layer 1: attention blend + rmsnorm -> A; out = partial + A @ attn_w^T
    attn_stage_kernel<<<ROWS, 256, 0, stream>>>(Vc, Pb, qn1, anw, Abuf);
    gemm_f16_wmma_kernel<<<ggrid, 256, 0, stream>>>(Abuf, W1h, Pb, out);

    // sub-layer 2: V[8] = out (updated partial); out = out + A @ mlp_w^T
    attn_stage_kernel<<<ROWS, 256, 0, stream>>>(Vc, out, qn2, mnw, Abuf);
    gemm_f16_wmma_kernel<<<ggrid, 256, 0, stream>>>(Abuf, W2h, out, out);
}
